// dijetReinforceLayer_23716809409214
// MI455X (gfx1250) — compile-verified, matrix operand order checked by reading, not run on hardware
//
#include <hip/hip_runtime.h>
#include <hip/hip_bf16.h>

// CDNA5 WMMA vector types
typedef __attribute__((ext_vector_type(16))) __bf16 v16bf;
typedef __attribute__((ext_vector_type(8)))  float  v8f;

#define ND     256
#define KS     3
#define NX     2
#define NPAIR  6
#define KTOT   (ND * KS)        // 768
#define BM     64               // rows (n,p) per block
#define BN     64               // output columns per block
#define KSTEP  96               // 3 WMMA k-steps per stage (lcm with KS=3)
#define NKC    (KTOT / KSTEP)   // 8 stages

// Workspace blob per (otile 0..3, kchunk 0..7): bf16[2][3][4][32][16] = 24576 B
#define BBLOB      24576
#define HL_STRIDE  12288        // bytes between hi-plane and lo-plane
#define KS_STRIDE  4096         // bytes per k-step within a plane
#define SUB_STRIDE 1024         // bytes per 16-row/col sub-tile
#define ABASE      0            // A region offset in LDS arena
#define BBASE      24576        // B region offset in LDS arena

// ---------------------------------------------------------------------------
// Prep: convert W (f32) once into hi/lo bf16, pre-swizzled into the exact
// 16x32 WMMA B-fragment layout:  g=kk32/8, lane=r+16*(g&1), slot=(kk32&7)+8*(g>>1)
// ---------------------------------------------------------------------------
__global__ __launch_bounds__(256)
void prep_W_kernel(const float* __restrict__ W, unsigned short* __restrict__ ws)
{
    int gid = blockIdx.x * 256 + threadIdx.x;      // 0 .. ND*KTOT-1
    int o   = gid / KTOT;
    int kk  = gid - o * KTOT;
    float v = W[gid];
    __bf16 h = (__bf16)v;
    __bf16 l = (__bf16)(v - (float)h);

    int ot  = o >> 6;
    int ol  = o & 63;
    int kc  = kk / KSTEP;
    int kl  = kk - kc * KSTEP;
    int ks  = kl >> 5;
    int k32 = kl & 31;
    int g   = k32 >> 3;
    int lane = (ol & 15) + 16 * (g & 1);
    int slot = (k32 & 7) + 8 * (g >> 1);
    int sub  = ol >> 4;

    size_t base = (size_t)(ot * NKC + kc) * (BBLOB / 2);   // in bf16 elements
    size_t eidx = (((size_t)ks * 4 + sub) * 32 + lane) * 16 + slot;
    ws[base + eidx]                           = __builtin_bit_cast(unsigned short, h);
    ws[base + (size_t)(HL_STRIDE / 2) + eidx] = __builtin_bit_cast(unsigned short, l);
}

// ---------------------------------------------------------------------------
// Main GEMM: out[m=(n,p), o] = sum_kk A[m,kk]*W[o,kk] + b[o]
// A[m,3c+k] = (k<2) ? x[n,c,2p+k] : d[n,c,p]
// split-bf16: c += ah*bh + ah*bl + al*bh  (~fp32 accuracy)
// ---------------------------------------------------------------------------
__global__ __launch_bounds__(256)
void dijet_wmma_kernel(const float* __restrict__ x,
                       const float* __restrict__ d,
                       const unsigned short* __restrict__ wsB,
                       const float* __restrict__ bias,
                       float* __restrict__ out)
{
    // Single LDS arena (sole shared object => LDS offset 0):
    //   [0,      24576) : A fragments  bf16[2][3][4][32][16]
    //   [24576,  49152) : B fragments  bf16[2][3][4][32][16]
    __shared__ __align__(32) char smem[2 * BBLOB];

    const int tid      = threadIdx.x;
    const int lane     = tid & 31;
    const int wid      = tid >> 5;
    const int wave_row = wid >> 1;    // 0..3 -> A sub-tile
    const int wave_col = wid & 1;     // 0..1 -> B sub-tiles 2wc, 2wc+1

    const int o0 = blockIdx.x * BN;   // fast dim => A tile reused via L2
    const int m0 = blockIdx.y * BM;

    // -------- loop-invariant A staging geometry --------
    const int cp  = tid & 15;         // c-pair: local c = 2cp, 2cp+1
    const int r16 = tid >> 4;         // row within 16-row sub-tile
    // byte offsets (hi plane, sub=0) of the 3 packed slot-pairs for kk=6cp..6cp+5
    int aoff[3];
    #pragma unroll
    for (int j = 0; j < 3; ++j) {
        int kkp = 6 * cp + 2 * j;     // even => aligned slot pair
        int ks  = kkp >> 5;
        int k32 = kkp & 31;
        int g   = k32 >> 3;
        int ln  = r16 + 16 * (g & 1);
        int sl  = (k32 & 7) + 8 * (g >> 1);
        aoff[j] = ks * KS_STRIDE + ln * 32 + sl * 2;
    }
    // per-e (sub-tile = e) source base offsets; advance by constants per stage
    int xbase[4], dbase[4];
    #pragma unroll
    for (int e = 0; e < 4; ++e) {
        int ml = r16 + e * 16;
        int m  = m0 + ml;
        int n  = m / NPAIR;
        int p  = m - n * NPAIR;
        xbase[e] = (n * ND + 2 * cp) * (NPAIR * NX) + p * NX;
        dbase[e] = (n * ND + 2 * cp) * NPAIR + p;
    }

    v8f acc0 = {}, acc1 = {};
    const int ocol0 = o0 + (2 * wave_col + 0) * 16 + (lane & 15);
    const int ocol1 = o0 + (2 * wave_col + 1) * 16 + (lane & 15);

    // Per-thread async-copy source (byte address) and LDS destination offset
    const unsigned long long wsTileAddr =
        (unsigned long long)(const char*)(wsB + (size_t)blockIdx.x * NKC * (BBLOB / 2))
        + (unsigned)tid * 96u;
    const unsigned ldsB = BBASE + (unsigned)tid * 96u;

    for (int kc = 0; kc < NKC; ++kc) {
        // ---- batched global loads: A sources (registers, no LDS dep yet) ----
        float2 xa[4], xb[4];
        float  da[4], db[4];
        #pragma unroll
        for (int e = 0; e < 4; ++e) {
            int xo  = xbase[e] + kc * (32 * NPAIR * NX);   // c0 advances 32/stage
            int do_ = dbase[e] + kc * (32 * NPAIR);
            xa[e] = *(const float2*)(x + xo);                  // c = 2cp   : k=0,1
            xb[e] = *(const float2*)(x + xo + NPAIR * NX);     // c = 2cp+1 : k=0,1
            da[e] = d[do_];                                    // c = 2cp   : k=2
            db[e] = d[do_ + NPAIR];                            // c = 2cp+1 : k=2
        }

        __syncthreads();   // previous stage's consumers done

        // ---- B: async DMA global -> LDS (ASYNCcnt path), 96 B per thread ----
        {
            unsigned long long gsrc = wsTileAddr + (unsigned)kc * (unsigned)BBLOB;
            #pragma unroll
            for (int i = 0; i < 6; ++i) {
                unsigned           ldst = ldsB + (unsigned)(i * 16);
                unsigned long long ga   = gsrc + (unsigned)(i * 16);
                asm volatile("global_load_async_to_lds_b128 %0, %1, off"
                             :: "v"(ldst), "v"(ga) : "memory");
            }
        }

        // ---- A: split f32 -> bf16 hi/lo, packed b32 stores (overlaps DMA) ----
        #pragma unroll
        for (int e = 0; e < 4; ++e) {
            float v6[6] = { xa[e].x, xa[e].y, da[e], xb[e].x, xb[e].y, db[e] };
            #pragma unroll
            for (int j = 0; j < 3; ++j) {
                float  v0 = v6[2 * j], v1 = v6[2 * j + 1];
                __bf16 h0 = (__bf16)v0, h1 = (__bf16)v1;
                __bf16 l0 = (__bf16)(v0 - (float)h0);
                __bf16 l1 = (__bf16)(v1 - (float)h1);
                unsigned hp = (unsigned)__builtin_bit_cast(unsigned short, h0)
                            | ((unsigned)__builtin_bit_cast(unsigned short, h1) << 16);
                unsigned lp = (unsigned)__builtin_bit_cast(unsigned short, l0)
                            | ((unsigned)__builtin_bit_cast(unsigned short, l1) << 16);
                char* dst = smem + ABASE + aoff[j] + e * SUB_STRIDE;
                *(unsigned*)(dst)             = hp;
                *(unsigned*)(dst + HL_STRIDE) = lp;
            }
        }

        // own-wave async copies complete, then block-wide visibility via barrier
        asm volatile("s_wait_asynccnt 0x0" ::: "memory");
        __syncthreads();

        // ---- consume: 3 k-steps x (2 tiles x 3 split terms) = 18 WMMA ----
        #pragma unroll
        for (int ks = 0; ks < 3; ++ks) {
            const char* Ab = smem + ABASE + ks * KS_STRIDE + wave_row * SUB_STRIDE + lane * 32;
            const char* Bb = smem + BBASE + ks * KS_STRIDE + (2 * wave_col) * SUB_STRIDE + lane * 32;
            v16bf ah  = *(const v16bf*)(Ab);
            v16bf al  = *(const v16bf*)(Ab + HL_STRIDE);
            v16bf b0h = *(const v16bf*)(Bb);
            v16bf b0l = *(const v16bf*)(Bb + HL_STRIDE);
            v16bf b1h = *(const v16bf*)(Bb + SUB_STRIDE);
            v16bf b1l = *(const v16bf*)(Bb + HL_STRIDE + SUB_STRIDE);

            acc0 = __builtin_amdgcn_wmma_f32_16x16x32_bf16(false, ah, false, b0h, (short)0, acc0, false, false);
            acc0 = __builtin_amdgcn_wmma_f32_16x16x32_bf16(false, ah, false, b0l, (short)0, acc0, false, false);
            acc0 = __builtin_amdgcn_wmma_f32_16x16x32_bf16(false, al, false, b0h, (short)0, acc0, false, false);
            acc1 = __builtin_amdgcn_wmma_f32_16x16x32_bf16(false, ah, false, b1h, (short)0, acc1, false, false);
            acc1 = __builtin_amdgcn_wmma_f32_16x16x32_bf16(false, ah, false, b1l, (short)0, acc1, false, false);
            acc1 = __builtin_amdgcn_wmma_f32_16x16x32_bf16(false, al, false, b1h, (short)0, acc1, false, false);
        }
    }

    const float bia0 = bias[ocol0];
    const float bia1 = bias[ocol1];

    // C/D layout: VGPR i, lanes 0-15 -> M=i, lanes 16-31 -> M=8+i; N = lane%16
    #pragma unroll
    for (int i = 0; i < 8; ++i) {
        int mt = i + 8 * (lane >> 4);
        int m  = m0 + wave_row * 16 + mt;
        int n  = m / NPAIR;
        int p  = m - n * NPAIR;
        out[(n * ND + ocol0) * NPAIR + p] = acc0[i] + bia0;
        out[(n * ND + ocol1) * NPAIR + p] = acc1[i] + bia1;
    }
}

extern "C" void kernel_launch(void* const* d_in, const int* in_sizes, int n_in,
                              void* d_out, int out_size, void* d_ws, size_t ws_size,
                              hipStream_t stream) {
    (void)n_in; (void)out_size; (void)ws_size;
    const float* x    = (const float*)d_in[0];  // [N, ND, 12]
    const float* d    = (const float*)d_in[1];  // [N, ND, 6]
    const float* W    = (const float*)d_in[2];  // [ND, ND, 3]
    const float* bias = (const float*)d_in[3];  // [ND]
    float* out = (float*)d_out;                 // [N, ND, 6]
    unsigned short* ws = (unsigned short*)d_ws; // 768 KB pre-swizzled W (hi/lo bf16)

    const int nbatch = in_sizes[1] / (ND * NPAIR);
    const int mtot   = nbatch * NPAIR;

    // 1) pre-swizzle W into fragment-ready hi/lo bf16 blobs (L2-resident)
    prep_W_kernel<<<dim3((ND * KTOT) / 256), dim3(256), 0, stream>>>(W, ws);

    // 2) GEMM; column tiles on fast grid dim for L2 reuse of the A stream
    dim3 grid(ND / BN, mtot / BM);   // (4, 1536)
    dijet_wmma_kernel<<<grid, dim3(256), 0, stream>>>(x, d, ws, bias, out);
}
